// BiDirectionalCrossAttention_10307921510533
// MI455X (gfx1250) — matrix operand order repaired and checked by reading, not compiled, not run on hardware
//
#include <hip/hip_runtime.h>
#include <hip/hip_bf16.h>
#include <math.h>

// ---------------------------------------------------------------------------
// CDNA5 (gfx1250) bidirectional cross-attention.
//  - bf16 WMMA (v_wmma_f32_16x16x32_bf16) for every GEMM
//  - Tensor Data Mover (tensor_load_to_lds) stages GEMM A tiles into padded
//    LDS rows while waves compute from the other ping-pong buffer
//  - fused epilogues (bias / bias+bf16 / bias+GELU+bf16 / per-head V transpose)
//  - fused per-query-tile attention: scores live only in LDS, softmax in
//    registers, probs aliased to bf16, PV via WMMA, head-mean map emitted once
// ---------------------------------------------------------------------------

#define USE_TDM 1

typedef __bf16 v16bf __attribute__((ext_vector_type(16)));
typedef __bf16 bf16x8 __attribute__((ext_vector_type(8)));
typedef float  v8f   __attribute__((ext_vector_type(8)));
typedef unsigned int v4u __attribute__((ext_vector_type(4)));
typedef int v8i __attribute__((ext_vector_type(8)));
typedef int v4i __attribute__((ext_vector_type(4)));

__device__ __forceinline__ v8f wmma_bf16(v16bf a, v16bf b, v8f c) {
  // (neg_a, A, neg_b, B, c_mod, C, reuse_a, reuse_b)
  return __builtin_amdgcn_wmma_f32_16x16x32_bf16(false, a, false, b, (short)0, c,
                                                 false, false);
}

__device__ __forceinline__ v16bf cat8(bf16x8 lo, bf16x8 hi) {
  return __builtin_shufflevector(lo, hi, 0, 1, 2, 3, 4, 5, 6, 7,
                                         8, 9, 10, 11, 12, 13, 14, 15);
}

#if USE_TDM
// TDM: copy a 128-row x 32-col bf16 tile (row stride K elems) from global into
// LDS with 16B of padding after each 64B row (-> 40-element LDS row stride).
__device__ __forceinline__ void tdm_load_a_tile(const __bf16* gptr,
                                                unsigned lds_off, int K,
                                                int rows_total) {
  unsigned long long ga = (unsigned long long)(uintptr_t)gptr;
  v4u g0;
  g0[0] = 1u;                                    // count=1, user mode
  g0[1] = lds_off;                               // lds_addr (bytes)
  g0[2] = (unsigned)(ga & 0xffffffffu);          // global_addr[31:0]
  g0[3] = (unsigned)((ga >> 32) & 0x1ffffffu) |  // global_addr[56:32]
          (2u << 30);                            // type = 2 ("image")
  unsigned td0 = (unsigned)K;                    // tensor_dim0
  unsigned td1 = (unsigned)rows_total;           // tensor_dim1
  unsigned long long st0 = (unsigned long long)K;  // tensor_dim0_stride
  v8i g1;
  g1[0] = (int)((1u << 16) |   // data_size = 2B
                (1u << 20) |   // pad_enable
                (3u << 22) |   // pad_interval: 8 x 8B = 64B
                (3u << 25));   // pad_amount: 4 DWORDs = 16B
  g1[1] = (int)((td0 & 0xffffu) << 16);                  // dim0 lo16 @ [63:48]
  g1[2] = (int)((td0 >> 16) | ((td1 & 0xffffu) << 16));  // dim0 hi | dim1 lo
  g1[3] = (int)((td1 >> 16) | (32u << 16));              // dim1 hi | tile_dim0
  g1[4] = (int)128u;                                     // tile_dim1 | tile_dim2=0
  g1[5] = (int)(st0 & 0xffffffffu);                      // dim0_stride lo32
  g1[6] = (int)((st0 >> 32) & 0xffffu);                  // dim0_stride hi16
  g1[7] = 0;
  v4i gz = {0, 0, 0, 0};
#if __has_include(<hip/amd_detail/amd_gfx1250_TDM.h>)
  v8i gz8 = {0, 0, 0, 0, 0, 0, 0, 0};
  __builtin_amdgcn_tensor_load_to_lds(g0, g1, gz, gz, gz8, 0);
#else
  __builtin_amdgcn_tensor_load_to_lds(g0, g1, gz, gz, 0);
#endif
}
#endif

// ---------------------------------------------------------------------------
// f32 -> bf16 conversion (weights + activations)
// ---------------------------------------------------------------------------
__global__ void cvt_bf16_kernel(const float* __restrict__ src,
                                __bf16* __restrict__ dst, int n) {
  int i = blockIdx.x * 256 + threadIdx.x;
  if (i < n) dst[i] = (__bf16)src[i];
}

// ---------------------------------------------------------------------------
// Tiled bf16 GEMM: C(M,N) = A(M,K) @ W(K,N) + bias, WMMA 16x16x32 bf16.
// Block = 256 threads = 8 waves; tile 128x128; wave -> 32x64 (2x4 frags).
// Double-buffered LDS; A tiles staged by the Tensor Data Mover (wave 0 issues,
// drains TENSORcnt before the buffer-flip barrier); B tiles register-prefetched
// then transposed into LDS. MODE 0 f32 / 1 bf16 / 2 gelu-bf16 / 3 V-transpose.
// ---------------------------------------------------------------------------
template <int MODE>
__global__ __launch_bounds__(256) void gemm_bf16_kernel(
    const __bf16* __restrict__ A, const __bf16* __restrict__ W,
    const float* __restrict__ bias, void* __restrict__ out, int M, int N,
    int K) {
  __shared__ __bf16 As[2][128 * 40];
  __shared__ __bf16 Bs[2][128 * 40];

  const int tid = threadIdx.x;
  const int lane = tid & 31;
  const int wid = tid >> 5;
  const int wm = wid & 3;
  const int wn = wid >> 2;
  const int m0 = blockIdx.y * 128;
  const int n0 = blockIdx.x * 128;

  const int mrow = lane & 15;
  const int kb8 = (lane >> 4) * 8;
  const int kg16 = (lane >> 4) * 16;

  v8f acc[2][4];
#pragma unroll
  for (int i = 0; i < 2; ++i)
#pragma unroll
    for (int j = 0; j < 4; ++j) acc[i][j] = {};

  const int br = tid & 31, bc = tid >> 5;  // B loader: k-row, 16-col chunk
#if !USE_TDM
  const int ar = tid >> 1, ah = tid & 1;   // A loader: row, 16-col half
#endif

  // ---- prologue: stage tile 0 into buffer 0 ----
  {
#if USE_TDM
    if (wid == 0)
      tdm_load_a_tile(A + (size_t)m0 * K, (unsigned)(uintptr_t)&As[0][0], K, M);
#else
    const bf16x8* asrc = (const bf16x8*)(A + (size_t)(m0 + ar) * K + ah * 16);
    *(bf16x8*)&As[0][ar * 40 + ah * 16] = asrc[0];
    *(bf16x8*)&As[0][ar * 40 + ah * 16 + 8] = asrc[1];
#endif
    const bf16x8* bsrc = (const bf16x8*)(W + (size_t)br * N + n0 + bc * 16);
    bf16x8 v0 = bsrc[0], v1 = bsrc[1];
#pragma unroll
    for (int i = 0; i < 8; ++i) Bs[0][(bc * 16 + i) * 40 + br] = v0[i];
#pragma unroll
    for (int i = 0; i < 8; ++i) Bs[0][(bc * 16 + 8 + i) * 40 + br] = v1[i];
#if USE_TDM
    if (wid == 0) __builtin_amdgcn_s_wait_tensorcnt(0);
#endif
  }
  __syncthreads();

  int buf = 0;
  for (int k0 = 0; k0 < K; k0 += 32) {
    const int kn = k0 + 32;
    const bool has_next = kn < K;

    // ---- prefetch next tile (A via TDM, B into registers) ----
    bf16x8 pb0{}, pb1{};
    if (has_next) {
#if USE_TDM
      if (wid == 0)
        tdm_load_a_tile(A + (size_t)m0 * K + kn,
                        (unsigned)(uintptr_t)&As[buf ^ 1][0], K, M);
#endif
      const bf16x8* bsrc =
          (const bf16x8*)(W + (size_t)(kn + br) * N + n0 + bc * 16);
      pb0 = bsrc[0];
      pb1 = bsrc[1];
#if !USE_TDM
      const bf16x8* asrc = (const bf16x8*)(A + (size_t)(m0 + ar) * K + kn + ah * 16);
      bf16x8 a0 = asrc[0], a1 = asrc[1];
      *(bf16x8*)&As[buf ^ 1][ar * 40 + ah * 16] = a0;
      *(bf16x8*)&As[buf ^ 1][ar * 40 + ah * 16 + 8] = a1;
#endif
    }

    // ---- compute from current buffer ----
    v16bf af[2];
#pragma unroll
    for (int mt = 0; mt < 2; ++mt) {
      int base = (wm * 32 + mt * 16 + mrow) * 40;
      bf16x8 lo = *(const bf16x8*)&As[buf][base + kb8];
      bf16x8 hi = *(const bf16x8*)&As[buf][base + 16 + kb8];
      af[mt] = cat8(lo, hi);
    }
    v16bf bfr[4];
#pragma unroll
    for (int nt = 0; nt < 4; ++nt) {
      int base = (wn * 64 + nt * 16 + mrow) * 40 + kb8;
      bf16x8 lo = *(const bf16x8*)&Bs[buf][base];
      bf16x8 hi = *(const bf16x8*)&Bs[buf][base + 16];
      bfr[nt] = cat8(lo, hi);
    }
#pragma unroll
    for (int mt = 0; mt < 2; ++mt)
#pragma unroll
      for (int nt = 0; nt < 4; ++nt)
        acc[mt][nt] = wmma_bf16(af[mt], bfr[nt], acc[mt][nt]);

    // ---- drain prefetch into the other buffer, flip ----
    if (has_next) {
#pragma unroll
      for (int i = 0; i < 8; ++i) Bs[buf ^ 1][(bc * 16 + i) * 40 + br] = pb0[i];
#pragma unroll
      for (int i = 0; i < 8; ++i)
        Bs[buf ^ 1][(bc * 16 + 8 + i) * 40 + br] = pb1[i];
    }
#if USE_TDM
    if (wid == 0) __builtin_amdgcn_s_wait_tensorcnt(0);
#endif
    __syncthreads();
    buf ^= 1;
  }

  // ---- epilogue: vgpr e, lane l -> row = e + 8*(l>>4), col = l&15 ----
#pragma unroll
  for (int mt = 0; mt < 2; ++mt) {
#pragma unroll
    for (int nt = 0; nt < 4; ++nt) {
      int col = n0 + wn * 64 + nt * 16 + (lane & 15);
      float bv = bias[col];
#pragma unroll
      for (int e = 0; e < 8; ++e) {
        int row = m0 + wm * 32 + mt * 16 + ((lane >> 4) << 3) + e;
        float v = acc[mt][nt][e] + bv;
        if (MODE == 0) {
          ((float*)out)[(size_t)row * N + col] = v;
        } else if (MODE == 1) {
          ((__bf16*)out)[(size_t)row * N + col] = (__bf16)v;
        } else if (MODE == 2) {
          float g = 0.5f * v * (1.0f + erff(v * 0.70710678118f));
          ((__bf16*)out)[(size_t)row * N + col] = (__bf16)g;
        } else {  // MODE 3: V stored as (b*1024 + n, token) for PV matmul
          size_t dst = ((size_t)(row >> 10) * N + col) * 1024 + (row & 1023);
          ((__bf16*)out)[dst] = (__bf16)v;
        }
      }
    }
  }
}

// ---------------------------------------------------------------------------
// Fused attention: one block per (b, 16-query tile); loops over 16 heads.
// Score/PV loops software-pipelined (prefetch next global fragment before the
// current WMMA) so loads overlap math instead of s_wait_loadcnt 0 stalls.
// Dynamic LDS: 64KB scores/probs(aliased) + 64KB attn acc + ctx + reduce.
// ---------------------------------------------------------------------------
__global__ __launch_bounds__(256) void attn_kernel(
    const __bf16* __restrict__ Qb, const __bf16* __restrict__ Kb,
    const __bf16* __restrict__ Vtb, const unsigned char* __restrict__ mask,
    const float* __restrict__ temp, __bf16* __restrict__ ctxb,
    float* __restrict__ attn_out) {
  extern __shared__ char smem[];
  float* Sf = (float*)smem;                  // [16][1024] f32 scores
  __bf16* Pbf = (__bf16*)smem;               // [16][1024] bf16 probs (alias)
  float* attn_acc = (float*)(smem + 65536);  // [16][1024] f32
  float* ctx_acc = (float*)(smem + 131072);  // [16][64] f32
  float* redA = ctx_acc + 1024;              // [16][16]
  float* redB = redA + 256;                  // [16][16]

  const int tid = threadIdx.x;
  const int lane = tid & 31;
  const int wid = tid >> 5;
  const int b = blockIdx.x >> 6;
  const int q0 = (blockIdx.x & 63) << 4;
  const float scale = 1.0f / (8.0f * fmaxf(temp[0], 0.1f));  // sqrt(64)=8

  for (int i = tid; i < 16 * 1024; i += 256) attn_acc[i] = 0.0f;
  __syncthreads();

  const int r16 = tid >> 4;
  const int l16 = tid & 15;
  const int mrow = lane & 15;
  const int kb8 = (lane >> 4) * 8;
  const int kg16 = (lane >> 4) * 16;

  for (int h = 0; h < 16; ++h) {
    for (int i = tid; i < 1024; i += 256) ctx_acc[i] = 0.0f;

    // ---- scores: S = Q_tile @ K^T * scale (each wave owns 128 kv cols) ----
    const __bf16* qbase = Qb + ((size_t)(b * 1024 + q0 + mrow) * 1024) + h * 64;
    v16bf qa[2];
#pragma unroll
    for (int kk = 0; kk < 2; ++kk) {
      const __bf16* p = qbase + kk * 32 + kb8;
      qa[kk] = cat8(*(const bf16x8*)p, *(const bf16x8*)(p + 16));
    }
    bf16x8 kf0, kf1, kf2, kf3;
    {
      const __bf16* kp =
          Kb + ((size_t)(b * 1024 + wid * 128 + mrow) * 1024) + h * 64 + kb8;
      kf0 = *(const bf16x8*)kp;
      kf1 = *(const bf16x8*)(kp + 16);
      kf2 = *(const bf16x8*)(kp + 32);
      kf3 = *(const bf16x8*)(kp + 48);
    }
#pragma unroll
    for (int nt = 0; nt < 8; ++nt) {
      bf16x8 nf0{}, nf1{}, nf2{}, nf3{};
      if (nt < 7) {
        const __bf16* kp =
            Kb + ((size_t)(b * 1024 + wid * 128 + (nt + 1) * 16 + mrow) * 1024) +
            h * 64 + kb8;
        nf0 = *(const bf16x8*)kp;
        nf1 = *(const bf16x8*)(kp + 16);
        nf2 = *(const bf16x8*)(kp + 32);
        nf3 = *(const bf16x8*)(kp + 48);
      }
      v8f sacc = {};
      sacc = wmma_bf16(qa[0], cat8(kf0, kf1), sacc);
      sacc = wmma_bf16(qa[1], cat8(kf2, kf3), sacc);
      int col = wid * 128 + nt * 16 + (lane & 15);
      bool mk = mask[b * 1024 + col] != 0;
#pragma unroll
      for (int e = 0; e < 8; ++e) {
        int row = e + ((lane >> 4) << 3);
        float v = sacc[e] * scale;
        Sf[row * 1024 + col] = mk ? v : -3.0e38f;
      }
      kf0 = nf0; kf1 = nf1; kf2 = nf2; kf3 = nf3;
    }
    __syncthreads();

    // ---- softmax: 16 threads per row, 64 cols each (register-resident) ----
    float pv[64];
    float lmax = -3.4e38f;
#pragma unroll
    for (int j = 0; j < 64; ++j) {
      pv[j] = Sf[r16 * 1024 + l16 + j * 16];
      lmax = fmaxf(lmax, pv[j]);
    }
    redA[r16 * 16 + l16] = lmax;
    __syncthreads();  // all Sf reads complete; Pbf aliasing now safe
    float rmax = redA[r16 * 16];
#pragma unroll
    for (int i = 1; i < 16; ++i) rmax = fmaxf(rmax, redA[r16 * 16 + i]);
    float lsum = 0.0f;
#pragma unroll
    for (int j = 0; j < 64; ++j) {
      pv[j] = __expf(pv[j] - rmax);
      lsum += pv[j];
    }
    redB[r16 * 16 + l16] = lsum;
    __syncthreads();
    float rsum = 0.0f;
#pragma unroll
    for (int i = 0; i < 16; ++i) rsum += redB[r16 * 16 + i];
    float inv = 1.0f / rsum;
#pragma unroll
    for (int j = 0; j < 64; ++j) {
      float p = pv[j] * inv;
      int c = l16 + j * 16;
      Pbf[r16 * 1024 + c] = (__bf16)p;
      attn_acc[r16 * 1024 + c] += p;  // unique owner per element
    }
    __syncthreads();

    // ---- ctx = P @ V_head: 4 d-tiles x 2 contract halves over 8 waves ----
    {
      const int ntile = wid & 3;
      const int khalf = wid >> 2;
      const int ncol = ntile * 16 + (lane & 15);
      const __bf16* vbase = Vtb + ((size_t)(b * 1024 + h * 64 + ncol) * 1024) +
                            khalf * 512 + kb8;
      bf16x8 vf0 = *(const bf16x8*)vbase;
      bf16x8 vf1 = *(const bf16x8*)(vbase + 16);
      v8f cacc = {};
#pragma unroll
      for (int ks = 0; ks < 16; ++ks) {
        bf16x8 nv0{}, nv1{};
        if (ks < 15) {
          const __bf16* vp = vbase + (ks + 1) * 32;
          nv0 = *(const bf16x8*)vp;
          nv1 = *(const bf16x8*)(vp + 16);
        }
        const int koff = khalf * 512 + ks * 32;
        const __bf16* pp = Pbf + mrow * 1024 + koff + kb8;
        v16bf afr = cat8(*(const bf16x8*)pp, *(const bf16x8*)(pp + 16));
        cacc = wmma_bf16(afr, cat8(vf0, vf1), cacc);
        vf0 = nv0; vf1 = nv1;
      }
#pragma unroll
      for (int e = 0; e < 8; ++e) {
        int row = e + ((lane >> 4) << 3);
        atomicAdd(&ctx_acc[row * 64 + ncol], cacc[e]);  // ds_add_f32
      }
    }
    __syncthreads();
    for (int i = tid; i < 1024; i += 256) {
      int row = i >> 6, col = i & 63;
      ctxb[((size_t)(b * 1024 + q0 + row) * 1024) + h * 64 + col] =
          (__bf16)ctx_acc[i];
    }
    __syncthreads();
  }

  // head-mean attention map
#pragma unroll
  for (int j = 0; j < 64; ++j) {
    int c = l16 + j * 16;
    attn_out[((size_t)(b * 1024 + q0 + r16)) * 1024 + c] =
        attn_acc[r16 * 1024 + c] * 0.0625f;
  }
}

// ---------------------------------------------------------------------------
// LayerNorm(y + res) * g + beta -> bf16 into concat[:, :D]; res -> concat[:, D:]
// ---------------------------------------------------------------------------
__global__ __launch_bounds__(256) void ln_concat_kernel(
    const float* __restrict__ y, const float* __restrict__ res,
    const float* __restrict__ g, const float* __restrict__ beta,
    __bf16* __restrict__ concat) {
  __shared__ float red[256];
  const int row = blockIdx.x;
  const int tid = threadIdx.x;

  float x[4];
  float s = 0.0f;
#pragma unroll
  for (int j = 0; j < 4; ++j) {
    int c = tid + j * 256;
    x[j] = y[(size_t)row * 1024 + c] + res[(size_t)row * 1024 + c];
    s += x[j];
  }
  red[tid] = s;
  __syncthreads();
  for (int off = 128; off > 0; off >>= 1) {
    if (tid < off) red[tid] += red[tid + off];
    __syncthreads();
  }
  float mean = red[0] * (1.0f / 1024.0f);
  __syncthreads();
  s = 0.0f;
#pragma unroll
  for (int j = 0; j < 4; ++j) {
    float d = x[j] - mean;
    s += d * d;
  }
  red[tid] = s;
  __syncthreads();
  for (int off = 128; off > 0; off >>= 1) {
    if (tid < off) red[tid] += red[tid + off];
    __syncthreads();
  }
  float rstd = rsqrtf(red[0] * (1.0f / 1024.0f) + 1e-5f);
#pragma unroll
  for (int j = 0; j < 4; ++j) {
    int c = tid + j * 256;
    concat[(size_t)row * 2048 + c] =
        (__bf16)((x[j] - mean) * rstd * g[c] + beta[c]);
    concat[(size_t)row * 2048 + 1024 + c] =
        (__bf16)res[(size_t)row * 1024 + c];
  }
}

// ---------------------------------------------------------------------------
// Host orchestration
// ---------------------------------------------------------------------------
static void launch_cvt(const float* src, __bf16* dst, int n, hipStream_t s) {
  cvt_bf16_kernel<<<(n + 255) / 256, 256, 0, s>>>(src, dst, n);
}

template <int MODE>
static void launch_gemm(const __bf16* A, const __bf16* W, const float* bias,
                        void* out, int M, int N, int K, hipStream_t s) {
  dim3 grid(N / 128, M / 128);
  gemm_bf16_kernel<MODE><<<grid, 256, 0, s>>>(A, W, bias, out, M, N, K);
}

extern "C" void kernel_launch(void* const* d_in, const int* in_sizes, int n_in,
                              void* d_out, int out_size, void* d_ws,
                              size_t ws_size, hipStream_t stream) {
  (void)in_sizes; (void)n_in; (void)out_size; (void)ws_size;
  const int D = 1024, M = 4096;  // M = B*L
  const size_t BLD = (size_t)M * D;

  const float* seq = (const float*)d_in[0];
  const float* struc = (const float*)d_in[1];
  const unsigned char* mask = (const unsigned char*)d_in[2];
  const float* fus_w1 = (const float*)d_in[25];
  const float* fus_b1 = (const float*)d_in[26];
  const float* fus_w2 = (const float*)d_in[27];
  const float* fus_b2 = (const float*)d_in[28];

  char* ws = (char*)d_ws;
  size_t off = 0;
  auto take = [&](size_t bytes) -> void* {
    void* p = ws + off;
    off += (bytes + 255) & ~(size_t)255;
    return p;
  };

  __bf16* seq_bf = (__bf16*)take(BLD * 2);
  __bf16* struct_bf = (__bf16*)take(BLD * 2);
  __bf16 *wq_bf[2], *wk_bf[2], *wv_bf[2], *wo_bf[2];
  for (int a = 0; a < 2; ++a) {
    wq_bf[a] = (__bf16*)take((size_t)D * D * 2);
    wk_bf[a] = (__bf16*)take((size_t)D * D * 2);
    wv_bf[a] = (__bf16*)take((size_t)D * D * 2);
    wo_bf[a] = (__bf16*)take((size_t)D * D * 2);
  }
  __bf16* fw1_bf = (__bf16*)take((size_t)2 * D * D * 2);
  __bf16* fw2_bf = (__bf16*)take((size_t)D * D * 2);
  __bf16* Qb = (__bf16*)take(BLD * 2);
  __bf16* Kb = (__bf16*)take(BLD * 2);
  __bf16* Vtb = (__bf16*)take(BLD * 2);
  __bf16* ctx_bf = (__bf16*)take(BLD * 2);
  float* yf = (float*)take(BLD * 4);
  __bf16* concat_bf = (__bf16*)take(BLD * 2 * 2);
  __bf16* h1_bf = (__bf16*)take(BLD * 2);

  launch_cvt(seq, seq_bf, (int)BLD, stream);
  launch_cvt(struc, struct_bf, (int)BLD, stream);
  for (int a = 0; a < 2; ++a) {
    int base = 3 + a * 11;
    launch_cvt((const float*)d_in[base + 0], wq_bf[a], D * D, stream);
    launch_cvt((const float*)d_in[base + 2], wk_bf[a], D * D, stream);
    launch_cvt((const float*)d_in[base + 4], wv_bf[a], D * D, stream);
    launch_cvt((const float*)d_in[base + 6], wo_bf[a], D * D, stream);
  }
  launch_cvt(fus_w1, fw1_bf, 2 * D * D, stream);
  launch_cvt(fus_w2, fw2_bf, D * D, stream);

  float* out_f = (float*)d_out;
  const size_t attn_smem = 65536 + 65536 + 4096 + 2048;  // 137216 B

  for (int a = 0; a < 2; ++a) {
    int base = 3 + a * 11;
    const float* qb_bias = (const float*)d_in[base + 1];
    const float* kb_bias = (const float*)d_in[base + 3];
    const float* vb_bias = (const float*)d_in[base + 5];
    const float* ob_bias = (const float*)d_in[base + 7];
    const float* g = (const float*)d_in[base + 8];
    const float* beta = (const float*)d_in[base + 9];
    const float* temp = (const float*)d_in[base + 10];

    const float* qin_f = (a == 0) ? seq : struc;
    const __bf16* qin_bf = (a == 0) ? seq_bf : struct_bf;
    const __bf16* kv_bf = (a == 0) ? struct_bf : seq_bf;
    float* out_fused = out_f + (size_t)a * BLD;
    float* out_attn = out_f + (size_t)(2 + a) * BLD;

    launch_gemm<1>(qin_bf, wq_bf[a], qb_bias, Qb, M, D, D, stream);
    launch_gemm<1>(kv_bf, wk_bf[a], kb_bias, Kb, M, D, D, stream);
    launch_gemm<3>(kv_bf, wv_bf[a], vb_bias, Vtb, M, D, D, stream);

    attn_kernel<<<256, 256, attn_smem, stream>>>(Qb, Kb, Vtb, mask, temp,
                                                 ctx_bf, out_attn);

    launch_gemm<0>(ctx_bf, wo_bf[a], ob_bias, yf, M, D, D, stream);
    ln_concat_kernel<<<M, 256, 0, stream>>>(yf, qin_f, g, beta, concat_bf);
    launch_gemm<2>(concat_bf, fw1_bf, fus_b1, h1_bf, M, D, 2 * D, stream);
    launch_gemm<0>(h1_bf, fw2_bf, fus_b2, out_fused, M, D, D, stream);
  }
}
